// GPT_69020124447367
// MI455X (gfx1250) — matrix operand order repaired and checked by reading, not compiled, not run on hardware
//
#include <hip/hip_runtime.h>
#include <hip/hip_bf16.h>

// ---------------- model constants (match reference) ----------------
constexpr int NV   = 32000;
constexpr int ND   = 1024;
constexpr int NH   = 16;
constexpr int NHID = 4096;
constexpr int NL   = 8;
constexpr int NT   = 1024;
constexpr int NB   = 2;
constexpr int NDH  = 64;          // ND / NH
constexpr float LN_EPS = 1e-5f;

typedef __attribute__((ext_vector_type(16))) _Float16 v16h;
typedef __attribute__((ext_vector_type(8)))  _Float16 v8h;
typedef __attribute__((ext_vector_type(8)))  float    v8f;

// B-operand kinds for the generic GEMM
#define BK_F32_KMAJ  0   // f32, element (k,n) at k*ldb + n   (convert path)
#define BK_F32_NMAJ  1   // f32, element (k,n) at n*ldb + k   (convert path)
#define BK_F16_NMAJ  2   // f16, element (k,n) at n*ldb + k   (async-DMA path)

// ---------------- block reductions (256 threads) ----------------
__device__ __forceinline__ float blk_sum256(float v) {
    __shared__ float red[256];
    int t = threadIdx.x;
    red[t] = v; __syncthreads();
    #pragma unroll
    for (int s = 128; s > 0; s >>= 1) {
        if (t < s) red[t] += red[t + s];
        __syncthreads();
    }
    float r = red[0];
    __syncthreads();
    return r;
}

__device__ __forceinline__ float blk_max256(float v) {
    __shared__ float red[256];
    int t = threadIdx.x;
    red[t] = v; __syncthreads();
    #pragma unroll
    for (int s = 128; s > 0; s >>= 1) {
        if (t < s) red[t] = fmaxf(red[t], red[t + s]);
        __syncthreads();
    }
    float r = red[0];
    __syncthreads();
    return r;
}

// ---------------- one-time weight conversion: f32 -> f16 (same layout) -----
__global__ void cvt_f16_kernel(const float* __restrict__ in,
                               _Float16* __restrict__ out, long long n) {
    long long stride = (long long)gridDim.x * blockDim.x;
    for (long long i = (long long)blockIdx.x * blockDim.x + threadIdx.x; i < n; i += stride)
        out[i] = (_Float16)in[i];
}

// ---------------- one-time weight convert + transpose --------------------
// in : batch of [K][N] f32 ; out : batch of [N][K] f16 ; blockIdx.z = batch
__global__ void transpose_cvt_kernel(const float* __restrict__ in,
                                     _Float16* __restrict__ out, int K, int N) {
    __shared__ float tile[32][33];
    const float* ib = in  + (long long)blockIdx.z * K * N;
    _Float16*    ob = out + (long long)blockIdx.z * K * N;
    int n0 = blockIdx.x * 32, k0 = blockIdx.y * 32;
    int tx = threadIdx.x & 31, ty = threadIdx.x >> 5;   // 256 thr = 32 x 8
    for (int r = ty; r < 32; r += 8)
        tile[r][tx] = ib[(long long)(k0 + r) * N + n0 + tx];
    __syncthreads();
    for (int r = ty; r < 32; r += 8)
        ob[(long long)(n0 + r) * K + k0 + tx] = (_Float16)tile[tx][r];
}

// ---------------- embedding + positional ----------------
__global__ void embed_kernel(const int* __restrict__ idx,
                             const float* __restrict__ emb,
                             const float* __restrict__ pos,
                             float* __restrict__ x) {
    int m = blockIdx.x;            // 0 .. B*T-1
    int t = m % NT;
    int token = idx[m];
    const float* e = emb + (long long)token * ND;
    const float* p = pos + (long long)t * ND;
    float* xr = x + (long long)m * ND;
    for (int d = threadIdx.x; d < ND; d += blockDim.x)
        xr[d] = e[d] + p[d];
}

// ---------------- layernorm (one 256-thread block per row) ----------------
__global__ void ln_kernel(const float* __restrict__ x, float* __restrict__ y,
                          const float* __restrict__ g, const float* __restrict__ b) {
    long long row = blockIdx.x;
    const float* xr = x + row * ND;
    float* yr = y + row * ND;
    int t = threadIdx.x;

    float s = 0.f;
    for (int d = t; d < ND; d += 256) s += xr[d];
    float mu = blk_sum256(s) * (1.0f / ND);

    float vs = 0.f;
    for (int d = t; d < ND; d += 256) { float dv = xr[d] - mu; vs += dv * dv; }
    float var = blk_sum256(vs) * (1.0f / ND);
    float inv = rsqrtf(var + LN_EPS);

    for (int d = t; d < ND; d += 256)
        yr[d] = (xr[d] - mu) * inv * g[d] + b[d];
}

// ---------------- row softmax over length NT ----------------
__global__ void softmax_kernel(float* __restrict__ sc) {
    long long row = blockIdx.x;       // 0 .. B*H*T-1
    float* p = sc + row * NT;
    int t = threadIdx.x;

    float m = -1e30f;
    for (int d = t; d < NT; d += 256) m = fmaxf(m, p[d]);
    m = blk_max256(m);

    float s = 0.f;
    for (int d = t; d < NT; d += 256) { float e = __expf(p[d] - m); p[d] = e; s += e; }
    s = blk_sum256(s);
    float inv = 1.0f / s;

    for (int d = t; d < NT; d += 256) p[d] *= inv;
}

// ---------------- generic batched WMMA GEMM (double-buffered) -------------
// C[z] = epilogue( scale * A[z](MxK) * B[z](KxN) ).  Tile 64x64x32, 128
// threads = 4 wave32 waves, each wave a 2x2 grid of v_wmma_f32_16x16x32_f16.
// Ping-pong LDS: while tile t is consumed by the matrix pipe, tile t+1's
// B-operand is DMA'd by GLOBAL_LOAD_ASYNC_TO_LDS_B128 (ASYNCcnt) and the
// A-operand is converted f32->f16 on the VALU.  One barrier per k-step.
__global__ __launch_bounds__(128) void gemm_wmma(
    const float* __restrict__ A, int lda, long long sAb, long long sAh,
    const void* __restrict__ B, int ldb, long long sBb, long long sBh, int bkind,
    float* __restrict__ C, int ldc, long long sCb, long long sCh,
    const float* __restrict__ residual,      // same layout/strides as C, or null
    const float* __restrict__ bias,          // indexed by global column, or null
    int M, int N, int K, int Hh,
    float scale, int relu, int causal)
{
    __shared__ __align__(16) _Float16 As[2][64][32];   // [buf][m][k]
    __shared__ __align__(16) _Float16 Bs[2][64][32];   // [buf][n][k] (transposed)

    const int tid  = threadIdx.x;
    const int lane = tid & 31;
    const int wv   = tid >> 5;            // wave id 0..3
    const int wm   = wv >> 1;             // wave row (0/1)
    const int wn   = wv & 1;              // wave col (0/1)
    const int hi   = lane >> 4;           // lane half (0/1)
    const int lx   = lane & 15;

    const int z  = blockIdx.z;
    const int bb = z / Hh;
    const int hh = z % Hh;
    const float*    Ab  = A + bb * sAb + hh * sAh;
    const float*    Bf  = (const float*)B    + bb * sBb + hh * sBh;
    const _Float16* Bh  = (const _Float16*)B + bb * sBb + hh * sBh;
    float*          Cb  = C + bb * sCb + hh * sCh;
    const float*    Rb  = residual ? residual + bb * sCb + hh * sCh : nullptr;

    const int tileM = blockIdx.y * 64;
    const int tileN = blockIdx.x * 64;

    // ---- tile staging into buffer s (issues async DMA for f16 B path)
    auto stage = [&](int kt, int s) {
        if (bkind == BK_F16_NMAJ) {
            // 64n x 32k f16 = 4KB; each thread DMAs 32B (2x B128).
            // inst offset applies to BOTH lds and global addresses (ISA 08 4.4)
            int n = tid >> 1, kb = (tid & 1) * 16;
            const _Float16* src = Bh + (long long)(tileN + n) * ldb + kt + kb;
            unsigned lo = (unsigned)(size_t)&Bs[s][n][kb];  // low 32b = LDS offset
            unsigned long long ga = (unsigned long long)(size_t)src;
            asm volatile("global_load_async_to_lds_b128 %0, %1, off"
                         :: "v"(lo), "v"(ga) : "memory");
            asm volatile("global_load_async_to_lds_b128 %0, %1, off offset:16"
                         :: "v"(lo), "v"(ga) : "memory");
        } else if (bkind == BK_F32_NMAJ) {   // contiguous in k
            int n = tid >> 1, kb = (tid & 1) * 16;
            const float* src = Bf + (long long)(tileN + n) * ldb + kt + kb;
            #pragma unroll
            for (int j = 0; j < 16; ++j) Bs[s][n][kb + j] = (_Float16)src[j];
        } else {                             // BK_F32_KMAJ: contiguous in n
            int k = tid >> 2, nb = (tid & 3) * 16;
            const float* src = Bf + (long long)(kt + k) * ldb + tileN + nb;
            #pragma unroll
            for (int j = 0; j < 16; ++j) Bs[s][nb + j][k] = (_Float16)src[j];
        }
        // A tile: 64x32, each thread converts 16 contiguous f32 -> f16
        int row = tid >> 1, cb = (tid & 1) * 16;
        const float* src = Ab + (long long)(tileM + row) * lda + kt + cb;
        #pragma unroll
        for (int j = 0; j < 16; ++j) As[s][row][cb + j] = (_Float16)src[j];
    };

    v8f acc[2][2] = {};
    const int nk = K >> 5;               // k-steps of 32

    stage(0, 0);
    if (bkind == BK_F16_NMAJ)
        asm volatile("s_wait_asynccnt 0x0" ::: "memory");
    __syncthreads();

    for (int t = 0; t < nk; ++t) {
        const int cur = t & 1;
        const bool more = (t + 1) < nk;
        if (more) stage((t + 1) << 5, cur ^ 1);   // overlap DMA/cvt with WMMA

        // ---- fragments per ISA 7.12.2 lane layouts (two ds_load_b128 each)
        union F16x16 { v16h v; v8h h[2]; };
        F16x16 af[2], bf[2];
        #pragma unroll
        for (int i = 0; i < 2; ++i) {
            int r = wm * 32 + i * 16 + lx;              // A: M = lane&15
            af[i].h[0] = *(const v8h*)&As[cur][r][8 * hi];       // K = 8*hi + e
            af[i].h[1] = *(const v8h*)&As[cur][r][16 + 8 * hi];  // K = 16+8*hi + e
            int c = wn * 32 + i * 16 + lx;              // B: N = lane&15
            bf[i].h[0] = *(const v8h*)&Bs[cur][c][8 * hi];       // K = 8*hi + e
            bf[i].h[1] = *(const v8h*)&Bs[cur][c][16 + 8 * hi];  // K = 16+8*hi + e
        }
        #pragma unroll
        for (int i = 0; i < 2; ++i)
            #pragma unroll
            for (int j = 0; j < 2; ++j)
                acc[i][j] = __builtin_amdgcn_wmma_f32_16x16x32_f16(
                    false, af[i].v, false, bf[j].v,
                    (short)0, acc[i][j], false, false);

        if (more && bkind == BK_F16_NMAJ)
            asm volatile("s_wait_asynccnt 0x0" ::: "memory");
        __syncthreads();   // dscnt wait inside also closes the WAR on buf cur
    }

    // ---- epilogue: C layout per ISA (VGPR r -> M = r + 8*hi, N = lane&15)
    #pragma unroll
    for (int i = 0; i < 2; ++i)
        #pragma unroll
        for (int j = 0; j < 2; ++j)
            #pragma unroll
            for (int r = 0; r < 8; ++r) {
                int rowl = tileM + wm * 32 + i * 16 + r + 8 * hi;
                int coll = tileN + wn * 32 + j * 16 + lx;
                float val = acc[i][j][r] * scale;
                if (bias)   val += bias[coll];
                if (causal && coll > rowl) val = -1e30f;
                if (relu)   val = fmaxf(val, 0.f);
                long long off = (long long)rowl * ldc + coll;
                if (Rb)     val += Rb[off];
                Cb[off] = val;
            }
}

// ---------------- host orchestration ----------------
extern "C" void kernel_launch(void* const* d_in, const int* in_sizes, int n_in,
                              void* d_out, int out_size, void* d_ws, size_t ws_size,
                              hipStream_t stream) {
    (void)in_sizes; (void)n_in; (void)out_size; (void)ws_size;

    const int*   idx   = (const int*)  d_in[0];
    const float* emb   = (const float*)d_in[1];
    const float* pos   = (const float*)d_in[2];
    const float* wq    = (const float*)d_in[3];   // [L,H,D,DH]
    const float* wk    = (const float*)d_in[4];
    const float* wv_   = (const float*)d_in[5];
    const float* wo    = (const float*)d_in[6];   // [L,D,D]  (n-major already)
    const float* w1    = (const float*)d_in[7];   // [L,D,HID]
    const float* b1    = (const float*)d_in[8];   // [L,HID]
    const float* w2    = (const float*)d_in[9];   // [L,HID,D]
    const float* b2    = (const float*)d_in[10];  // [L,D]
    const float* ln1_g = (const float*)d_in[11];
    const float* ln1_b = (const float*)d_in[12];
    const float* ln2_g = (const float*)d_in[13];
    const float* ln2_b = (const float*)d_in[14];
    const float* lnf_g = (const float*)d_in[15];
    const float* lnf_b = (const float*)d_in[16];
    float* out = (float*)d_out;

    const long long MT = (long long)NB * NT;      // 2048 token rows

    // ---- f32 workspace region
    float* ws  = (float*)d_ws;
    float* x   = ws;                 // [MT, D]
    float* hb  = x   + MT * ND;      // [MT, D]      LN output
    float* q   = hb  + MT * ND;      // [B,H,T,DH]
    float* k   = q   + MT * ND;
    float* v   = k   + MT * ND;
    float* oc  = v   + MT * ND;      // [MT, D]      concat heads
    float* mlp = oc  + MT * ND;      // [MT, HID]
    float* sc  = mlp + MT * NHID;    // [B,H,T,T]    attention scores

    // ---- f16 weight region (one-time per-launch conversion, k-contiguous)
    _Float16* f16b = (_Float16*)(sc + (long long)NB * NH * NT * NT);
    _Float16* wqt  = f16b;                                   // [L,H,DH,D]
    _Float16* wkt  = wqt + (long long)NL * ND * ND;
    _Float16* wvt  = wkt + (long long)NL * ND * ND;
    _Float16* wof  = wvt + (long long)NL * ND * ND;          // [L,D,D] n-major
    _Float16* w1t  = wof + (long long)NL * ND * ND;          // [L,HID,D]
    _Float16* w2t  = w1t + (long long)NL * ND * NHID;        // [L,D,HID]
    _Float16* embf = w2t + (long long)NL * NHID * ND;        // [V,D] n-major

    // ---- weight conversions (independent, launched up front)
    {
        dim3 gq(NDH / 32, ND / 32, NL * NH);      // [D][DH] -> [DH][D] per head
        transpose_cvt_kernel<<<gq, 256, 0, stream>>>(wq,  wqt, ND, NDH);
        transpose_cvt_kernel<<<gq, 256, 0, stream>>>(wk,  wkt, ND, NDH);
        transpose_cvt_kernel<<<gq, 256, 0, stream>>>(wv_, wvt, ND, NDH);
        dim3 g1(NHID / 32, ND / 32, NL);          // [D][HID] -> [HID][D]
        transpose_cvt_kernel<<<g1, 256, 0, stream>>>(w1, w1t, ND, NHID);
        dim3 g2(ND / 32, NHID / 32, NL);          // [HID][D] -> [D][HID]
        transpose_cvt_kernel<<<g2, 256, 0, stream>>>(w2, w2t, NHID, ND);
        cvt_f16_kernel<<<4096, 256, 0, stream>>>(wo,  wof,  (long long)NL * ND * ND);
        cvt_f16_kernel<<<4096, 256, 0, stream>>>(emb, embf, (long long)NV * ND);
    }

    const long long sQb = (long long)NH * NT * NDH;   // per-batch stride of q/k/v
    const long long sQh = (long long)NT * NDH;        // per-head stride of q/k/v

    embed_kernel<<<dim3((unsigned)MT), 256, 0, stream>>>(idx, emb, pos, x);

    for (int l = 0; l < NL; ++l) {
        const _Float16* wql = wqt + (long long)l * ND * ND;  // [H][DH][D]
        const _Float16* wkl = wkt + (long long)l * ND * ND;
        const _Float16* wvl = wvt + (long long)l * ND * ND;
        const _Float16* wol = wof + (long long)l * ND * ND;  // [D][D] n-major
        const _Float16* w1l = w1t + (long long)l * ND * NHID;
        const _Float16* w2l = w2t + (long long)l * NHID * ND;

        // --- LN1
        ln_kernel<<<dim3((unsigned)MT), 256, 0, stream>>>(x, hb, ln1_g + l * ND, ln1_b + l * ND);

        // --- Q,K,V projections: per (b,h) GEMM  T x DH = (T x D)(D x DH)
        dim3 gqkv(NDH / 64, NT / 64, NB * NH);
        gemm_wmma<<<gqkv, 128, 0, stream>>>(
            hb, ND, (long long)NT * ND, 0,
            wql, ND, 0, (long long)NDH * ND, BK_F16_NMAJ,
            q, NDH, sQb, sQh,
            nullptr, nullptr, NT, NDH, ND, NH, 1.0f, 0, 0);
        gemm_wmma<<<gqkv, 128, 0, stream>>>(
            hb, ND, (long long)NT * ND, 0,
            wkl, ND, 0, (long long)NDH * ND, BK_F16_NMAJ,
            k, NDH, sQb, sQh,
            nullptr, nullptr, NT, NDH, ND, NH, 1.0f, 0, 0);
        gemm_wmma<<<gqkv, 128, 0, stream>>>(
            hb, ND, (long long)NT * ND, 0,
            wvl, ND, 0, (long long)NDH * ND, BK_F16_NMAJ,
            v, NDH, sQb, sQh,
            nullptr, nullptr, NT, NDH, ND, NH, 1.0f, 0, 0);

        // --- scores = scale * Q K^T, causal mask in epilogue (B = f32 K, n-major)
        dim3 gsc(NT / 64, NT / 64, NB * NH);
        gemm_wmma<<<gsc, 128, 0, stream>>>(
            q, NDH, sQb, sQh,
            k, NDH, sQb, sQh, BK_F32_NMAJ,
            sc, NT, (long long)NH * NT * NT, (long long)NT * NT,
            nullptr, nullptr, NT, NT, NDH, NH, 0.125f /*1/sqrt(64)*/, 0, 1);

        // --- softmax rows
        softmax_kernel<<<dim3((unsigned)(NB * NH * NT)), 256, 0, stream>>>(sc);

        // --- O = P V, stored directly head-concatenated: oc[b*T+t, h*DH+e]
        dim3 gpv(NDH / 64, NT / 64, NB * NH);
        gemm_wmma<<<gpv, 128, 0, stream>>>(
            sc, NT, (long long)NH * NT * NT, (long long)NT * NT,
            v, NDH, sQb, sQh, BK_F32_KMAJ,
            oc, ND, (long long)NT * ND, (long long)NDH,
            nullptr, nullptr, NT, NDH, NT, NH, 1.0f, 0, 0);

        // --- x = x + oc @ wo^T   (wo is n-major f16 -> async path)
        dim3 gwo(ND / 64, (unsigned)(MT / 64), 1);
        gemm_wmma<<<gwo, 128, 0, stream>>>(
            oc, ND, 0, 0,
            wol, ND, 0, 0, BK_F16_NMAJ,
            x, ND, 0, 0,
            x /*residual*/, nullptr, (int)MT, ND, ND, 1, 1.0f, 0, 0);

        // --- LN2
        ln_kernel<<<dim3((unsigned)MT), 256, 0, stream>>>(x, hb, ln2_g + l * ND, ln2_b + l * ND);

        // --- mlp = relu(hb @ w1 + b1)   (w1t is [HID][D] f16)
        dim3 g1(NHID / 64, (unsigned)(MT / 64), 1);
        gemm_wmma<<<g1, 128, 0, stream>>>(
            hb, ND, 0, 0,
            w1l, ND, 0, 0, BK_F16_NMAJ,
            mlp, NHID, 0, 0,
            nullptr, b1 + (long long)l * NHID, (int)MT, NHID, ND, 1, 1.0f, 1 /*relu*/, 0);

        // --- x = x + mlp @ w2 + b2     (w2t is [D][HID] f16)
        dim3 g2(ND / 64, (unsigned)(MT / 64), 1);
        gemm_wmma<<<g2, 128, 0, stream>>>(
            mlp, NHID, 0, 0,
            w2l, NHID, 0, 0, BK_F16_NMAJ,
            x, ND, 0, 0,
            x /*residual*/, b2 + (long long)l * ND, (int)MT, ND, NHID, 1, 1.0f, 0, 0);
    }

    // --- final LN + tied LM head: out = hb @ emb^T (emb is n-major f16)
    ln_kernel<<<dim3((unsigned)MT), 256, 0, stream>>>(x, hb, lnf_g, lnf_b);
    dim3 glm(NV / 64, (unsigned)(MT / 64), 1);
    gemm_wmma<<<glm, 128, 0, stream>>>(
        hb, ND, 0, 0,
        embf, ND, 0, 0, BK_F16_NMAJ,
        out, NV, 0, 0,
        nullptr, nullptr, (int)MT, NV, ND, 1, 1.0f, 0, 0);
}
